// GIN_GNN_75677323755666
// MI455X (gfx1250) — matrix operand order repaired and checked by reading, not compiled, not run on hardware
//
#include <hip/hip_runtime.h>
#include <hip/hip_bf16.h>
#include <math.h>

#define HDIM 128
#define ELU_A 0.1f

typedef __attribute__((ext_vector_type(16))) _Float16 v16h;
typedef __attribute__((ext_vector_type(8)))  float    v8f;

union CV { v8f v; float f[8]; };

// ---------------- utility kernels ----------------

__global__ void fill_kernel(float* __restrict__ p, float val, int n) {
    int i = blockIdx.x * blockDim.x + threadIdx.x;
    if (i < n) p[i] = val;
}

// zero only the first `dim` columns of an [N, HDIM] buffer
__global__ void zero2d_kernel(float* __restrict__ p, int N, int dim) {
    int i = blockIdx.x * blockDim.x + threadIdx.x;
    if (i >= N * dim) return;
    int r = i / dim, c = i - r * dim;
    p[r * HDIM + c] = 0.0f;
}

// Convert f32 weights [srcRows x 128] into WMMA-B-fragment-swizzled f16.
// dst flat index i = ((kt*8 + nt)*32 + lane)*16 + j
//   lane = nl + 16*hh ; element j of lane's v16h corresponds to
//   k = kt*32 + hh*8 + (j<8 ? j : j+8) ; n = nt*16 + nl
__global__ void cvt_swizzle_kernel(const float* __restrict__ src, _Float16* __restrict__ dst,
                                   int srcRows, int Kp) {
    int i = blockIdx.x * blockDim.x + threadIdx.x;
    if (i >= Kp * HDIM) return;
    int j    = i & 15;
    int lane = (i >> 4) & 31;
    int nt   = (i >> 9) & 7;
    int kt   = i >> 12;
    int hh = lane >> 4, nl = lane & 15;
    int k = kt * 32 + hh * 8 + (j < 8 ? j : j + 8);
    int n = nt * 16 + nl;
    dst[i] = (k < srcRows) ? (_Float16)src[k * HDIM + n] : (_Float16)0.0f;
}

__global__ void concat_kernel(const float* __restrict__ h0, const float* __restrict__ co,
                              float* __restrict__ X, int N) {
    int i = blockIdx.x * blockDim.x + threadIdx.x;
    if (i >= N * 16) return;
    int node = i >> 4, c = i & 15;
    X[node * HDIM + c] = (c < 13) ? h0[node * 13 + c] : co[node * 3 + (c - 13)];
}

// dim/4 lanes per edge, float4 gather + 4 scalar atomics per lane
__global__ void scatter_kernel(const float* __restrict__ X, const int* __restrict__ ei,
                               float* __restrict__ AGG, int E, int dim) {
    int lpe = dim >> 2;                       // lanes per edge (4 or 32)
    int gt = blockIdx.x * blockDim.x + threadIdx.x;
    int edge = gt / lpe;
    if (edge >= E) return;
    int q = gt - edge * lpe;
    int c0 = q * 4;
    int s = ei[edge];
    int d = ei[E + edge];
    float4 v = *(const float4*)&X[s * HDIM + c0];
    float* ag = &AGG[d * HDIM + c0];
    atomicAdd(ag + 0, v.x);
    atomicAdd(ag + 1, v.y);
    atomicAdd(ag + 2, v.z);
    atomicAdd(ag + 3, v.w);
}

__device__ inline void atomicMaxF(float* addr, float val) {
    unsigned int* ua = (unsigned int*)addr;
    unsigned int old = *ua;
    while (__uint_as_float(old) < val) {
        unsigned int assumed = old;
        old = atomicCAS(ua, assumed, __float_as_uint(val));
        if (old == assumed) break;
    }
}

// ---------------- WMMA fragment staging helper ----------------
// Store f32 value for (row m, col k) into LDS at the A-fragment position:
// lane = m + 16*hh, element j, where k = kt*32 + ka and
// ka = (j<8 ? j : j+8) + hh*8  (ISA 16-bit A-matrix layout)
__device__ inline void stage_a(_Float16* sAf /*[kt][32][16]*/, int m, int k, float v) {
    int kt = k >> 5;
    int kk = k & 31;
    int hh = (kk >> 3) & 1;
    int base = kk - 8 * hh;                 // in {0..7} u {16..23}
    int j = (base < 8) ? base : (base - 8); // 0..15
    sAf[(kt * 32 + m + 16 * hh) * 16 + j] = (_Float16)v;
}

// ---------------- WMMA GEMM kernels ----------------
// Block: 256 threads = 8 waves; computes rows [m0, m0+16) x all 128 cols.
// Wave w owns columns [16w, 16w+16). A staged pre-swizzled in LDS.

template <int KP>
__global__ __launch_bounds__(256)
void gemm_mlp1(const float* __restrict__ X, const float* __restrict__ AGG,
               const _Float16* __restrict__ Wh, const float* __restrict__ bias,
               float* __restrict__ Z, float* __restrict__ stats, int K, int N) {
    __shared__ alignas(32) _Float16 sAf[(KP / 32) * 32 * 16];
    const int m0 = blockIdx.x * 16;
    const int tid = threadIdx.x;
    for (int idx = tid; idx < 16 * KP; idx += 256) {
        int r = idx / KP, c = idx - r * KP;
        float v = 0.0f;
        if (c < K && (m0 + r) < N) {
            int g = (m0 + r) * HDIM + c;
            v = X[g] + AGG[g];
        }
        stage_a(sAf, r, c, v);
    }
    __syncthreads();

    const int wave = tid >> 5;
    const int lane = tid & 31;
    const int hh   = lane >> 4;
    const int nl   = lane & 15;
    const int n0   = wave * 16;

    CV cv;
    float bn = bias[n0 + nl];
    #pragma unroll
    for (int r = 0; r < 8; ++r) cv.f[r] = bn;

    const v16h* Wv = (const v16h*)Wh;
    __builtin_prefetch(Wh, 0, 0);
    #pragma unroll
    for (int kt = 0; kt < KP / 32; ++kt) {
        v16h av = *(const v16h*)&sAf[(kt * 32 + lane) * 16];
        v16h bv = Wv[(kt * 8 + wave) * 32 + lane];
        cv.v = __builtin_amdgcn_wmma_f32_16x16x32_f16(false, av, false, bv,
                                                      (short)0, cv.v, false, false);
    }

    float lsum = 0.0f, lsq = 0.0f;
    float* zp = &Z[(m0 + hh * 8) * HDIM + n0 + nl];
    if (m0 + 16 <= N) {                       // block-uniform fast path
        #pragma unroll
        for (int r = 0; r < 8; ++r) {
            float v = cv.f[r];
            zp[r * HDIM] = v;
            lsum += v; lsq += v * v;
        }
    } else {
        #pragma unroll
        for (int r = 0; r < 8; ++r) {
            if (m0 + hh * 8 + r < N) {
                float v = cv.f[r];
                zp[r * HDIM] = v;
                lsum += v; lsq += v * v;
            }
        }
    }
    atomicAdd(&stats[n0 + nl], lsum);
    atomicAdd(&stats[HDIM + n0 + nl], lsq);
}

__global__ __launch_bounds__(256)
void gemm_mlp2(const float* __restrict__ Z, const float* __restrict__ ss,
               const _Float16* __restrict__ Wh, const float* __restrict__ bias,
               float* __restrict__ Xout, int N) {
    __shared__ alignas(32) _Float16 sAf[4 * 32 * 16];
    const int m0 = blockIdx.x * 16;
    const int tid = threadIdx.x;
    for (int idx = tid; idx < 16 * HDIM; idx += 256) {
        int r = idx >> 7, c = idx & 127;
        float v = 0.0f;
        if ((m0 + r) < N) {
            v = Z[(m0 + r) * HDIM + c] * ss[c] + ss[HDIM + c];  // batchnorm
            v = v > 0.0f ? v : 0.0f;                            // relu
        }
        stage_a(sAf, r, c, v);
    }
    __syncthreads();

    const int wave = tid >> 5;
    const int lane = tid & 31;
    const int hh   = lane >> 4;
    const int nl   = lane & 15;
    const int n0   = wave * 16;

    CV cv;
    float bn = bias[n0 + nl];
    #pragma unroll
    for (int r = 0; r < 8; ++r) cv.f[r] = bn;

    const v16h* Wv = (const v16h*)Wh;
    __builtin_prefetch(Wh, 0, 0);
    #pragma unroll
    for (int kt = 0; kt < 4; ++kt) {
        v16h av = *(const v16h*)&sAf[(kt * 32 + lane) * 16];
        v16h bv = Wv[(kt * 8 + wave) * 32 + lane];
        cv.v = __builtin_amdgcn_wmma_f32_16x16x32_f16(false, av, false, bv,
                                                      (short)0, cv.v, false, false);
    }

    float* xp = &Xout[(m0 + hh * 8) * HDIM + n0 + nl];
    if (m0 + 16 <= N) {
        #pragma unroll
        for (int r = 0; r < 8; ++r) {
            float v = cv.f[r];
            v = v > 0.0f ? v : ELU_A * (__expf(v) - 1.0f);      // elu
            xp[r * HDIM] = v;
        }
    } else {
        #pragma unroll
        for (int r = 0; r < 8; ++r) {
            if (m0 + hh * 8 + r < N) {
                float v = cv.f[r];
                v = v > 0.0f ? v : ELU_A * (__expf(v) - 1.0f);
                xp[r * HDIM] = v;
            }
        }
    }
}

// ---------------- BN finalize, pooling, classifier ----------------

__global__ void bn_finalize_kernel(const float* __restrict__ stats,
                                   const float* __restrict__ gamma,
                                   const float* __restrict__ beta,
                                   float* __restrict__ ss, float invN) {
    int t = threadIdx.x;   // 128 threads
    float mean = stats[t] * invN;
    float var  = stats[HDIM + t] * invN - mean * mean;
    float sc   = gamma[t] * rsqrtf(var + 1e-5f);
    ss[t]        = sc;
    ss[HDIM + t] = beta[t] - mean * sc;
}

__global__ void pool_kernel(const float* __restrict__ X, const int* __restrict__ batch,
                            float* __restrict__ PM, float* __restrict__ PX,
                            float* __restrict__ CNT, int N) {
    int gt = blockIdx.x * blockDim.x + threadIdx.x;
    int node = gt >> 5, lane = gt & 31;
    if (node >= N) return;
    int b = batch[node];
    if (lane == 0) atomicAdd(&CNT[b], 1.0f);
    int c0 = lane * 4;
    float4 v = *(const float4*)&X[node * HDIM + c0];
    float* pm = &PM[b * HDIM + c0];
    float* px = &PX[b * HDIM + c0];
    atomicAdd(pm + 0, v.x); atomicMaxF(px + 0, v.x);
    atomicAdd(pm + 1, v.y); atomicMaxF(px + 1, v.y);
    atomicAdd(pm + 2, v.z); atomicMaxF(px + 2, v.z);
    atomicAdd(pm + 3, v.w); atomicMaxF(px + 3, v.w);
}

__global__ __launch_bounds__(128)
void classifier_kernel(const float* __restrict__ PM, const float* __restrict__ PX,
                       const float* __restrict__ CNT, const float* __restrict__ g0,
                       const float* __restrict__ Wc1, const float* __restrict__ bc1,
                       const float* __restrict__ Wc2, const float* __restrict__ bc2,
                       float* __restrict__ out, int GF) {
    __shared__ float xg[2 * HDIM + 8];
    __shared__ float hid[HDIM];
    const int b = blockIdx.x;
    const int t = threadIdx.x;
    float denom = fmaxf(CNT[b], 1.0f);
    xg[t]        = PM[b * HDIM + t] / denom;
    xg[HDIM + t] = PX[b * HDIM + t];
    if (t < GF) xg[2 * HDIM + t] = g0[b * GF + t];
    __syncthreads();
    const int KD = 2 * HDIM + GF;
    float acc = bc1[t];
    for (int k = 0; k < KD; ++k) acc += xg[k] * Wc1[k * HDIM + t];
    hid[t] = acc > 0.0f ? acc : ELU_A * (__expf(acc) - 1.0f);
    __syncthreads();
    if (t == 0) {
        float l0 = bc2[0], l1 = bc2[1];
        for (int k = 0; k < HDIM; ++k) {
            l0 += hid[k] * Wc2[k * 2 + 0];
            l1 += hid[k] * Wc2[k * 2 + 1];
        }
        float m = fmaxf(l0, l1);
        float e0 = __expf(l0 - m), e1 = __expf(l1 - m);
        float s = e0 + e1;
        out[b * 2 + 0] = e0 / s;
        out[b * 2 + 1] = e1 / s;
    }
}

// ---------------- host orchestration ----------------

extern "C" void kernel_launch(void* const* d_in, const int* in_sizes, int n_in,
                              void* d_out, int out_size, void* d_ws, size_t ws_size,
                              hipStream_t stream) {
    const float* h0     = (const float*)d_in[0];
    const float* coord0 = (const float*)d_in[1];
    const float* g0     = (const float*)d_in[2];
    const int*   eidx   = (const int*)  d_in[3];
    const int*   batch  = (const int*)  d_in[4];
    const float* W1_0   = (const float*)d_in[5];
    const float* b1_0   = (const float*)d_in[6];
    const float* g_0    = (const float*)d_in[7];
    const float* be_0   = (const float*)d_in[8];
    const float* W2_0   = (const float*)d_in[9];
    const float* b2_0   = (const float*)d_in[10];
    const float* Ws1    = (const float*)d_in[11];
    const float* bs1    = (const float*)d_in[12];
    const float* gs     = (const float*)d_in[13];
    const float* bes    = (const float*)d_in[14];
    const float* Ws2    = (const float*)d_in[15];
    const float* bs2    = (const float*)d_in[16];
    const float* Wc1    = (const float*)d_in[17];
    const float* bc1    = (const float*)d_in[18];
    const float* Wc2    = (const float*)d_in[19];
    const float* bc2    = (const float*)d_in[20];

    const int N = in_sizes[0] / 13;
    const int E = in_sizes[3] / 2;
    const int B = in_sizes[2] / 8;
    const int GF = 8;

    float* ws = (float*)d_ws;
    size_t o = 0;
    float* X     = ws + o;  o += (size_t)N * HDIM;
    float* AGG   = ws + o;  o += (size_t)N * HDIM;
    float* Z     = ws + o;  o += (size_t)N * HDIM;
    float* STATS = ws + o;  o += 2 * HDIM;
    float* SS    = ws + o;  o += 2 * HDIM;
    float* PM    = ws + o;  o += (size_t)B * HDIM;
    float* PX    = ws + o;  o += (size_t)B * HDIM;
    float* CNT   = ws + o;  o += B;
    o = (o + 15) & ~(size_t)15;               // 64B-align the f16 region
    _Float16* WH = (_Float16*)(ws + o);
    _Float16* wA0 = WH;                       // 32x128 swizzled (padded from 16x128)
    _Float16* wB0 = wA0 + 32 * HDIM;          // 128x128 swizzled
    _Float16* wA1 = wB0 + HDIM * HDIM;
    _Float16* wB1 = wA1 + HDIM * HDIM;
    _Float16* wA2 = wB1 + HDIM * HDIM;
    _Float16* wB2 = wA2 + HDIM * HDIM;

    const int TB = 256;
    #define GR(n) (((n) + TB - 1) / TB)

    // weight conversion f32 -> f16 fragment-swizzled (W1_0 zero-padded K 16->32)
    cvt_swizzle_kernel<<<GR(32 * HDIM),   TB, 0, stream>>>(W1_0,              wA0, 16,  32);
    cvt_swizzle_kernel<<<GR(HDIM * HDIM), TB, 0, stream>>>(W2_0,              wB0, HDIM, HDIM);
    cvt_swizzle_kernel<<<GR(HDIM * HDIM), TB, 0, stream>>>(Ws1,               wA1, HDIM, HDIM);
    cvt_swizzle_kernel<<<GR(HDIM * HDIM), TB, 0, stream>>>(Ws2,               wB1, HDIM, HDIM);
    cvt_swizzle_kernel<<<GR(HDIM * HDIM), TB, 0, stream>>>(Ws1 + HDIM * HDIM, wA2, HDIM, HDIM);
    cvt_swizzle_kernel<<<GR(HDIM * HDIM), TB, 0, stream>>>(Ws2 + HDIM * HDIM, wB2, HDIM, HDIM);

    // x0 = concat(h0, coord0) into first 16 cols of X (stride 128)
    concat_kernel<<<GR(N * 16), TB, 0, stream>>>(h0, coord0, X, N);

    const _Float16* wA[3] = { wA0, wA1, wA2 };
    const _Float16* wB[3] = { wB0, wB1, wB2 };
    const float* B1[3] = { b1_0, bs1, bs1 + HDIM };
    const float* B2[3] = { b2_0, bs2, bs2 + HDIM };
    const float* GM[3] = { g_0,  gs,  gs  + HDIM };
    const float* BT[3] = { be_0, bes, bes + HDIM };
    const int    KD[3] = { 16, HDIM, HDIM };

    const int gemmBlocks = (N + 15) / 16;
    for (int L = 0; L < 3; ++L) {
        const int dim = KD[L];
        zero2d_kernel<<<GR(N * dim), TB, 0, stream>>>(AGG, N, dim);
        fill_kernel<<<1, TB, 0, stream>>>(STATS, 0.0f, 2 * HDIM);
        scatter_kernel<<<GR(E * (dim / 4)), TB, 0, stream>>>(X, eidx, AGG, E, dim);
        if (L == 0)
            gemm_mlp1<32><<<gemmBlocks, TB, 0, stream>>>(X, AGG, wA[L], B1[L], Z, STATS, dim, N);
        else
            gemm_mlp1<HDIM><<<gemmBlocks, TB, 0, stream>>>(X, AGG, wA[L], B1[L], Z, STATS, dim, N);
        bn_finalize_kernel<<<1, HDIM, 0, stream>>>(STATS, GM[L], BT[L], SS, 1.0f / (float)N);
        gemm_mlp2<<<gemmBlocks, TB, 0, stream>>>(Z, SS, wB[L], B2[L], X, N);
    }

    // pooling
    fill_kernel<<<GR(B * HDIM), TB, 0, stream>>>(PM, 0.0f, B * HDIM);
    fill_kernel<<<GR(B * HDIM), TB, 0, stream>>>(PX, -INFINITY, B * HDIM);
    fill_kernel<<<1, TB, 0, stream>>>(CNT, 0.0f, B);
    pool_kernel<<<GR(N * 32), TB, 0, stream>>>(X, batch, PM, PX, CNT, N);

    // classifier + softmax -> d_out [B, 2] f32
    classifier_kernel<<<B, HDIM, 0, stream>>>(PM, PX, CNT, g0, Wc1, bc1, Wc2, bc2,
                                              (float*)d_out, GF);
    #undef GR
}
